// SmoothMetricLossV2_26792005992920
// MI455X (gfx1250) — matrix operand order repaired and checked by reading, not compiled, not run on hardware
//
#include <hip/hip_runtime.h>
#include <hip/hip_bf16.h>
#include <math.h>

// SmoothMetricLoss (lifted structured loss) for MI455X / gfx1250, wave32.
// d = cdist(E,E) computed tile-by-tile via v_wmma_f32_16x16x32_bf16 with a
// hi/lo bf16 split of fp32 (3 WMMAs per k-step ~ fp32 accuracy), staged into
// a double-buffered LDS pipeline with GLOBAL_LOAD_ASYNC_TO_LDS_B128
// (ASYNCcnt), and consumed by a fused sqrt/exp/mask/row-reduce epilogue.
// d never touches HBM; the only large HBM stream is the 64MB mask (prefetched).

#define TWO_N  8192
#define DDIM   256
#define NHALF  4096
#define MARGIN_C 0.5f

#define TM 128          // block tile rows
#define TN 64           // block tile cols
#define KC 32           // k-chunk (== WMMA K)
#define NCH (DDIM / KC) // 8 k-chunks
#define LDA 40          // LDS row stride in bf16: 80B = 5*16B aligned, 20-bank step

typedef __attribute__((ext_vector_type(16))) __bf16 v16bf;
typedef __attribute__((ext_vector_type(8)))  __bf16 v8bf;
typedef __attribute__((ext_vector_type(8)))  float  v8f;

union V16 { v16bf v; v8bf h[2]; };

// ---- CDNA5 async global->LDS copy (16B per lane), tracked by ASYNCcnt ------
__device__ __forceinline__ void async_ld16(void* lds_dst, const void* gsrc) {
    // operand order mirrors "global_load_b128 vdst, vaddr, off":
    // vdst = per-lane LDS byte address, vaddr = 64-bit global address
    unsigned int lds = (unsigned int)(unsigned long long)lds_dst;
    asm volatile("global_load_async_to_lds_b128 %0, %1, off"
                 :: "v"(lds), "v"(gsrc) : "memory");
}

// ---------------- pass 1: fp32 -> bf16 hi/lo split + squared row norms ------
__global__ __launch_bounds__(DDIM)
void prep_kernel(const float* __restrict__ E,
                 __bf16* __restrict__ Ehi, __bf16* __restrict__ Elo,
                 float* __restrict__ norms) {
    __shared__ float red[DDIM];
    const int row = blockIdx.x;
    const int t   = threadIdx.x;
    float x = E[(size_t)row * DDIM + t];
    __bf16 hi = (__bf16)x;
    __bf16 lo = (__bf16)(x - (float)hi);
    Ehi[(size_t)row * DDIM + t] = hi;
    Elo[(size_t)row * DDIM + t] = lo;
    red[t] = x * x;
    __syncthreads();
    for (int s = DDIM / 2; s > 0; s >>= 1) {
        if (t < s) red[t] += red[t + s];
        __syncthreads();
    }
    if (t == 0) norms[row] = red[0];
}

// ---------------- pass 2: WMMA Gram tiles + fused distance/exp epilogue -----
__global__ __launch_bounds__(256)
void tile_kernel(const __bf16* __restrict__ Ehi, const __bf16* __restrict__ Elo,
                 const float* __restrict__ norms,
                 const unsigned char* __restrict__ mask,
                 float* __restrict__ partial, float* __restrict__ ap) {
    // double-buffered stages: 2 * (A 2*10240B + B 2*5120B) = 61440 B
    __shared__ __bf16 lAhi[2][TM * LDA];
    __shared__ __bf16 lAlo[2][TM * LDA];
    __shared__ __bf16 lBhi[2][TN * LDA];
    __shared__ __bf16 lBlo[2][TN * LDA];

    const int t    = threadIdx.x;
    const int wave = t >> 5;          // 8 waves
    const int lane = t & 31;
    const int l15  = lane & 15;
    const int half = lane >> 4;       // 0 or 1
    const int wm   = wave >> 1;       // 0..3 : 32-row band
    const int wn   = wave & 1;        // 0..1 : 32-col band
    const int m0   = blockIdx.y * TM;
    const int n0   = blockIdx.x * TN;

    // Prefetch this tile's mask slab (consumed only in the epilogue) so its
    // HBM fetch overlaps the whole GEMM phase.
    if (t < TM)
        __builtin_prefetch(&mask[(size_t)(m0 + t) * TWO_N + n0], 0, 1);

    v8f acc[2][2] = {};               // [m-subtile][n-subtile] 16x16 f32 each

    // Async staging plan: 6 x 16B per thread per stage (uniform):
    //   A(hi,lo): row = t>>1 (0..127), 32B half c = (t&1)*32  -> 2+2 copies
    //   B(hi,lo): row = t>>2 (0..63),  16B chunk  (t&3)*16    -> 1+1 copies
    const int arow = t >> 1, acol = (t & 1) * 32;          // bytes within 64B row
    const int brow = t >> 2, bcol = (t & 3) * 16;
    const char* gAhi = (const char*)Ehi + (size_t)(m0 + arow) * (DDIM * 2) + acol;
    const char* gAlo = (const char*)Elo + (size_t)(m0 + arow) * (DDIM * 2) + acol;
    const char* gBhi = (const char*)Ehi + (size_t)(n0 + brow) * (DDIM * 2) + bcol;
    const char* gBlo = (const char*)Elo + (size_t)(n0 + brow) * (DDIM * 2) + bcol;

    auto issue_stage = [&](int st) {
        const int b = st & 1;
        const int kb = st * (KC * 2);                      // k byte offset
        char* dAhi = (char*)lAhi[b] + arow * (LDA * 2) + acol;
        char* dAlo = (char*)lAlo[b] + arow * (LDA * 2) + acol;
        async_ld16(dAhi,      gAhi + kb);
        async_ld16(dAhi + 16, gAhi + kb + 16);
        async_ld16(dAlo,      gAlo + kb);
        async_ld16(dAlo + 16, gAlo + kb + 16);
        async_ld16((char*)lBhi[b] + brow * (LDA * 2) + bcol, gBhi + kb);
        async_ld16((char*)lBlo[b] + brow * (LDA * 2) + bcol, gBlo + kb);
    };

    issue_stage(0);

    for (int s = 0; s < NCH; ++s) {
        if (s + 1 < NCH) {
            issue_stage(s + 1);                            // prefetch next stage
            asm volatile("s_wait_asynccnt 0x6" ::: "memory"); // stage s landed
        } else {
            asm volatile("s_wait_asynccnt 0x0" ::: "memory");
        }
        __syncthreads();

        const int b = s & 1;
        // Build operands per ISA 7.12.2 layouts.
        // A 16x32: lanes 0-15 hold K {0..7,16..23}, lanes 16-31 K {8..15,24..31}
        // B 32x16: lanes 0-15 hold K {0..7,16..23}, lanes 16-31 K {8..15,24..31}
        v16bf aH[2], aL[2], bH[2], bL[2];
        const int c0 = half * 8;
#pragma unroll
        for (int sb = 0; sb < 2; ++sb) {
            const __bf16* pa = lAhi[b] + (wm * 32 + sb * 16 + l15) * LDA;
            const __bf16* pl = lAlo[b] + (wm * 32 + sb * 16 + l15) * LDA;
            V16 u;
            u.h[0] = *(const v8bf*)(pa + c0);
            u.h[1] = *(const v8bf*)(pa + 16 + c0);
            aH[sb] = u.v;
            u.h[0] = *(const v8bf*)(pl + c0);
            u.h[1] = *(const v8bf*)(pl + 16 + c0);
            aL[sb] = u.v;
            const __bf16* pb  = lBhi[b] + (wn * 32 + sb * 16 + l15) * LDA + half * 8;
            const __bf16* pbl = lBlo[b] + (wn * 32 + sb * 16 + l15) * LDA + half * 8;
            u.h[0] = *(const v8bf*)(pb);
            u.h[1] = *(const v8bf*)(pb + 16);
            bH[sb] = u.v;
            u.h[0] = *(const v8bf*)(pbl);
            u.h[1] = *(const v8bf*)(pbl + 16);
            bL[sb] = u.v;
        }

#pragma unroll
        for (int ms = 0; ms < 2; ++ms)
#pragma unroll
            for (int ns = 0; ns < 2; ++ns) {
                // S += Ahi*Bhi + Ahi*Blo + Alo*Bhi   (fp32-split GEMM)
                acc[ms][ns] = __builtin_amdgcn_wmma_f32_16x16x32_bf16(
                    false, aH[ms], false, bH[ns], (short)0, acc[ms][ns], false, false);
                acc[ms][ns] = __builtin_amdgcn_wmma_f32_16x16x32_bf16(
                    false, aH[ms], false, bL[ns], (short)0, acc[ms][ns], false, false);
                acc[ms][ns] = __builtin_amdgcn_wmma_f32_16x16x32_bf16(
                    false, aL[ms], false, bH[ns], (short)0, acc[ms][ns], false, false);
            }
        __syncthreads();   // all waves done reading buf b before stage s+2 overwrites
    }

    // ---- fused epilogue: d = sqrt(x2+y2-2S), masked exp, ap band, row sums
    float y2v[2];
    int cv[2];
#pragma unroll
    for (int ns = 0; ns < 2; ++ns) {
        cv[ns] = n0 + wn * 32 + ns * 16 + l15;   // global column for this lane
        y2v[ns] = norms[cv[ns]];
    }

#pragma unroll
    for (int ms = 0; ms < 2; ++ms) {
#pragma unroll
        for (int j = 0; j < 8; ++j) {
            const int row = m0 + wm * 32 + ms * 16 + j + half * 8;  // C layout: VGPR j
            const float x2 = norms[row];
            float rsum = 0.0f;
#pragma unroll
            for (int ns = 0; ns < 2; ++ns) {
                float S  = acc[ms][ns][j];
                float sq = fmaxf(x2 + y2v[ns] - 2.0f * S, 0.0f);
                float d  = sqrtf(sq + 1e-12f);
                float v  = mask[(size_t)row * TWO_N + cv[ns]]
                               ? __expf(MARGIN_C - d) : 0.0f;
                rsum += v;
                if (row < NHALF && cv[ns] == row + NHALF) ap[row] = d;  // anchor-positive
            }
            // reduce over the 16 lanes of each half (N dimension)
#pragma unroll
            for (int o = 1; o < 16; o <<= 1) rsum += __shfl_xor(rsum, o, 32);
            if (l15 == 0)
                partial[(size_t)(blockIdx.x * 2 + wn) * TWO_N + row] = rsum;
        }
    }
}

// ---------------- pass 3: fold 256 partial column-slices per row ------------
__global__ __launch_bounds__(256)
void rowsum_kernel(const float* __restrict__ partial, float* __restrict__ row_sums) {
    const int r = blockIdx.x * 256 + threadIdx.x;
    float s = 0.0f;
    for (int p = 0; p < 2 * (TWO_N / TN); ++p)
        s += partial[(size_t)p * TWO_N + r];
    row_sums[r] = s;
}

// ---------------- pass 4: j = log(rs_a + rs_b) + ap, NaN-filtered mean/2 ----
__global__ __launch_bounds__(256)
void loss_kernel(const float* __restrict__ row_sums, const float* __restrict__ ap,
                 float* __restrict__ out) {
    __shared__ float sred[256];
    __shared__ int   cred[256];
    const int t = threadIdx.x;
    float s = 0.0f;
    int   c = 0;
    for (int i = t; i < NHALF; i += 256) {
        float rs = row_sums[i] + row_sums[i + NHALF];
        float j  = logf(rs) + ap[i];
        if (j == j) {  // !isnan
            float r = fmaxf(j, 0.0f);
            s += r * r;
            c += 1;
        }
    }
    sred[t] = s; cred[t] = c;
    __syncthreads();
    for (int st = 128; st > 0; st >>= 1) {
        if (t < st) { sred[t] += sred[t + st]; cred[t] += cred[t + st]; }
        __syncthreads();
    }
    if (t == 0) {
        int cnt = cred[0] > 0 ? cred[0] : 1;
        out[0] = sred[0] / (float)cnt * 0.5f;
    }
}

extern "C" void kernel_launch(void* const* d_in, const int* in_sizes, int n_in,
                              void* d_out, int out_size, void* d_ws, size_t ws_size,
                              hipStream_t stream) {
    const float*         E    = (const float*)d_in[0];
    const unsigned char* mask = (const unsigned char*)d_in[1];  // jax bool = 1 byte

    char* ws = (char*)d_ws;
    size_t off = 0;
    __bf16* Ehi = (__bf16*)(ws + off); off += (size_t)TWO_N * DDIM * sizeof(__bf16);
    __bf16* Elo = (__bf16*)(ws + off); off += (size_t)TWO_N * DDIM * sizeof(__bf16);
    float* norms    = (float*)(ws + off); off += (size_t)TWO_N * sizeof(float);
    float* ap       = (float*)(ws + off); off += (size_t)NHALF * sizeof(float);
    float* row_sums = (float*)(ws + off); off += (size_t)TWO_N * sizeof(float);
    float* partial  = (float*)(ws + off);
    off += (size_t)2 * (TWO_N / TN) * TWO_N * sizeof(float);
    // total workspace ~16.9 MB; every slot consumed is written each call
    // before it is read, so no zero-init needed (graph-replay safe).
    (void)ws_size; (void)in_sizes; (void)n_in; (void)out_size;

    prep_kernel<<<TWO_N, DDIM, 0, stream>>>(E, Ehi, Elo, norms);
    tile_kernel<<<dim3(TWO_N / TN, TWO_N / TM), 256, 0, stream>>>(
        Ehi, Elo, norms, mask, partial, ap);
    rowsum_kernel<<<TWO_N / 256, 256, 0, stream>>>(partial, row_sums);
    loss_kernel<<<1, 256, 0, stream>>>(row_sums, ap, (float*)d_out);
}
